// CharCountCNN_9466107920459
// MI455X (gfx1250) — matrix operand order repaired and checked by reading, not compiled, not run on hardware
//
#include <hip/hip_runtime.h>
#include <hip/hip_bf16.h>

typedef __attribute__((ext_vector_type(16))) __bf16 v16bf;
typedef __attribute__((ext_vector_type(8)))  float  v8f;

#define T_LEN 1024
#define B_LEN 128
#define NCLASS 80
#define NSTYLE 256
#define NCHSTYLE 64
#define CPAD1 416   // 400 padded to 13*32
#define TP (T_LEN + 2)   // halo-padded time dim: row 0 and row T+1 are zeros

// ---------------- argmax over class dim: chars[t*B+b] ----------------
__global__ __launch_bounds__(256) void argmax_kernel(
    const float* __restrict__ input, int* __restrict__ chars)
{
    int idx = blockIdx.x * blockDim.x + threadIdx.x; // t*B + b
    if (idx >= T_LEN * B_LEN) return;
    const float* p = input + (size_t)idx * NCLASS;
    float best = p[0]; int bi = 0;
    for (int c = 1; c < NCLASS; ++c) {
        float v = p[c];
        if (v > best) { best = v; bi = c; }
    }
    chars[idx] = bi;
}

// ---------------- zero the two halo rows of an activation buffer --------------
__global__ __launch_bounds__(256) void zero_halo_kernel(__bf16* __restrict__ x, int CPAD)
{
    int idx = blockIdx.x * blockDim.x + threadIdx.x;
    int total = B_LEN * CPAD * 2;
    if (idx >= total) return;
    int c  = idx % CPAD;
    int rb = idx / CPAD;
    int b  = rb >> 1;
    size_t row = (rb & 1) ? (size_t)(TP - 1) : 0;
    x[((size_t)b * TP + row) * CPAD + c] = (__bf16)0.0f;
}

// ---------------- pack conv weight (CO,CI,KT) fp32 -> WMMA A-fragment bf16 ----
// out layout: [tap][kc][mtile][lane][i], i fastest (16 bf16 contiguous per lane)
// A 16x32 bf16 fragment: m = lane%16, half = lane/16, k = half*16 + i
__global__ __launch_bounds__(256) void pack_w_kernel(
    const float* __restrict__ w, __bf16* __restrict__ out,
    int CO, int CI, int KT, int KC)
{
    int idx = blockIdx.x * blockDim.x + threadIdx.x;
    int mtiles = CO >> 4;
    int total = KT * KC * mtiles * 32 * 16;
    if (idx >= total) return;
    int i    = idx & 15;
    int lane = (idx >> 4) & 31;
    int rest = idx >> 9;
    int mtile = rest % mtiles; rest /= mtiles;
    int kc    = rest % KC;
    int tap   = rest / KC;
    int m    = mtile * 16 + (lane & 15);
    int half = lane >> 4;
    int k    = half * 16 + i;
    int ci   = kc * 32 + k;
    float v = (ci < CI) ? w[((size_t)m * CI + ci) * KT + tap] : 0.0f;
    out[idx] = (__bf16)v;
}

// ---------------- build concatenated input X[b][t+1][cp] (cp<416, bf16) ------
__global__ __launch_bounds__(256) void build_x_kernel(
    const float* __restrict__ input, const float* __restrict__ g_style,
    const float* __restrict__ char_style, const int* __restrict__ chars,
    __bf16* __restrict__ xblk)
{
    int idx = blockIdx.x * blockDim.x + threadIdx.x;
    if (idx >= B_LEN * T_LEN * CPAD1) return;
    int cp = idx % CPAD1;
    int t  = (idx / CPAD1) % T_LEN;
    int b  = idx / (CPAD1 * T_LEN);
    float v;
    if (cp < NCLASS) {
        v = input[((size_t)t * B_LEN + b) * NCLASS + cp];
    } else if (cp < NCLASS + NSTYLE) {
        v = g_style[(size_t)b * NSTYLE + (cp - NCLASS)];
    } else if (cp < NCLASS + NSTYLE + NCHSTYLE) {
        int ch = chars[t * B_LEN + b];
        v = char_style[((size_t)b * NCLASS + ch) * NCHSTYLE + (cp - NCLASS - NSTYLE)];
    } else {
        v = 0.0f;
    }
    xblk[((size_t)b * TP + (t + 1)) * CPAD1 + cp] = (__bf16)v;
}

// ---------------- WMMA conv-as-GEMM -------------------------------------------
// Y[b][co][t] = bias[co] + sum_{tap,ci} Wp * X[b][t+tap][ci]   (X halo-padded)
// X layout [B][TP][CPAD] bf16 -> direct B-fragment loads, no boundary branches.
// workgroup = 256 threads (8 waves) computes CO x 128t tile for batch b.
// Each wave computes an MW x NW block of 16x16 tiles.
template<int CO, int KC, int MW, int NW>
__global__ __launch_bounds__(256) void conv_gemm_kernel(
    const __bf16* __restrict__ xblk, const __bf16* __restrict__ wpack,
    const float* __restrict__ bias, float* __restrict__ y)
{
    constexpr int CPAD    = KC * 32;
    constexpr int COTILES = CO / 16;
    constexpr int COGRP   = COTILES / MW;  // wave groups along co
    static_assert(COGRP * (8 / NW) == 8, "wave mapping must cover 8 waves");

    int b    = blockIdx.y;
    int t0   = blockIdx.x * 128;
    int lane = threadIdx.x & 31;
    int w    = threadIdx.x >> 5;
    int cotile0 = (w % COGRP) * MW;
    int nt0     = (w / COGRP) * NW;
    int n     = lane & 15;
    int khalf = lane >> 4;

    v8f acc[MW][NW];
#pragma unroll
    for (int mi = 0; mi < MW; ++mi)
#pragma unroll
        for (int j = 0; j < NW; ++j)
#pragma unroll
            for (int q = 0; q < 8; ++q) acc[mi][j][q] = 0.0f;

    const __bf16* xb = xblk + (size_t)b * TP * CPAD;

    for (int tap = 0; tap < 3; ++tap) {
        // halo-padded: logical t + (tap-1) + 1 = t + tap
        int tbase = t0 + nt0 * 16 + n + tap;
        const __bf16* bbase = xb + (size_t)tbase * CPAD + khalf * 16;
        const __bf16* abase =
            wpack + ((size_t)tap * KC * COTILES + cotile0) * 512 + lane * 16;
#pragma unroll 2
        for (int kc = 0; kc < KC; ++kc) {
            v16bf af[MW];
#pragma unroll
            for (int mi = 0; mi < MW; ++mi)
                af[mi] = *(const v16bf*)(abase + ((size_t)kc * COTILES + mi) * 512);
#pragma unroll
            for (int j = 0; j < NW; ++j) {
                v16bf bfrag = *(const v16bf*)(bbase + kc * 32 + j * 16 * CPAD);
#pragma unroll
                for (int mi = 0; mi < MW; ++mi)
                    acc[mi][j] = __builtin_amdgcn_wmma_f32_16x16x32_bf16(
                        false, af[mi], false, bfrag, (short)0, acc[mi][j], false, false);
            }
        }
    }

    // D fragment: col n = lane%16, row m = v + 8*(lane>>4)
    float* yb = y + (size_t)b * CO * T_LEN;
#pragma unroll
    for (int mi = 0; mi < MW; ++mi)
#pragma unroll
        for (int j = 0; j < NW; ++j) {
            int t = t0 + (nt0 + j) * 16 + n;
#pragma unroll
            for (int v = 0; v < 8; ++v) {
                int co = (cotile0 + mi) * 16 + v + 8 * khalf;
                yb[(size_t)co * T_LEN + t] = acc[mi][j][v] + bias[co];
            }
        }
}

// ---------------- GroupNorm(32) + ReLU, fp32 [B][CO][T] -> bf16 [B][TP][CO] ---
template<int CO>
__global__ __launch_bounds__(256) void gn_relu_kernel(
    const float* __restrict__ y, __bf16* __restrict__ x,
    const float* __restrict__ scale, const float* __restrict__ bias)
{
    constexpr int CPG = CO / 32;
    constexpr int N   = CPG * T_LEN;
    int b = blockIdx.x >> 5;
    int g = blockIdx.x & 31;
    const float* yg = y + ((size_t)b * CO + g * CPG) * T_LEN;

    float sum = 0.f, sq = 0.f;
    for (int i = threadIdx.x; i < N; i += 256) {
        float v = yg[i];
        sum += v; sq += v * v;
    }
    __shared__ float s1[256], s2[256];
    s1[threadIdx.x] = sum; s2[threadIdx.x] = sq;
    __syncthreads();
    for (int st = 128; st > 0; st >>= 1) {
        if (threadIdx.x < st) {
            s1[threadIdx.x] += s1[threadIdx.x + st];
            s2[threadIdx.x] += s2[threadIdx.x + st];
        }
        __syncthreads();
    }
    float mu  = s1[0] / (float)N;
    float var = s2[0] / (float)N - mu * mu;
    float inv = rsqrtf(var + 1e-5f);

    for (int i = threadIdx.x; i < N; i += 256) {
        int c = g * CPG + i / T_LEN;
        int t = i % T_LEN;
        float v = (yg[i] - mu) * inv * scale[c] + bias[c];
        v = v > 0.f ? v : 0.f;
        x[((size_t)b * TP + (t + 1)) * CO + c] = (__bf16)v;
    }
}

// ---------------- conv4 (k=1, 32->1) + affine, out (T,B,1) --------------------
__global__ __launch_bounds__(256) void conv4_out_kernel(
    const __bf16* __restrict__ x3, const float* __restrict__ w4,
    const float* __restrict__ b4, const float* __restrict__ mean,
    const float* __restrict__ stdv, float* __restrict__ out)
{
    int idx = blockIdx.x * blockDim.x + threadIdx.x; // t*B + b
    if (idx >= T_LEN * B_LEN) return;
    int t = idx / B_LEN, b = idx % B_LEN;
    const __bf16* p = x3 + ((size_t)b * TP + (t + 1)) * 32;
    float acc = b4[0];
#pragma unroll
    for (int c = 0; c < 32; ++c) acc += (float)p[c] * w4[c];
    out[idx] = acc * stdv[0] + mean[0];
}

// ---------------- workspace layout (bytes) ------------------------------------
// xcat [B][TP][416] bf16 = 109,264,896 B; region later reused for X1/X2/X3
#define OFF_CHARS 0u                          // 512 KB
#define OFF_W1    524288u                     // 319,488 B
#define OFF_W2    843776u                     // 49,152 B
#define OFF_W3    892928u                     // 12,288 B
#define OFF_XR    1048576u
#define OFF_X1    (OFF_XR)                    // [B][TP][128] bf16 = 33,619,968 B
#define OFF_X2    (OFF_XR + 33619968u)        // [B][TP][64]  bf16 = 16,809,984 B
#define OFF_X3    (OFF_XR + 50429952u)        // [B][TP][32]  bf16 =  8,404,992 B
#define OFF_Y     (OFF_XR + 109264896u)       // 64 MB shared Y1/Y2/Y3 scratch

extern "C" void kernel_launch(void* const* d_in, const int* in_sizes, int n_in,
                              void* d_out, int out_size, void* d_ws, size_t ws_size,
                              hipStream_t stream)
{
    const float* input      = (const float*)d_in[0];
    const float* g_style    = (const float*)d_in[1];
    /* d_in[2] spacing_style unused */
    const float* char_style = (const float*)d_in[3];
    const float* w1 = (const float*)d_in[4];   const float* b1 = (const float*)d_in[5];
    const float* g1s = (const float*)d_in[6];  const float* g1b = (const float*)d_in[7];
    const float* w2 = (const float*)d_in[8];   const float* b2 = (const float*)d_in[9];
    const float* g2s = (const float*)d_in[10]; const float* g2b = (const float*)d_in[11];
    const float* w3 = (const float*)d_in[12];  const float* b3 = (const float*)d_in[13];
    const float* g3s = (const float*)d_in[14]; const float* g3b = (const float*)d_in[15];
    const float* w4 = (const float*)d_in[16];  const float* b4 = (const float*)d_in[17];
    const float* mean = (const float*)d_in[18];
    const float* stdv = (const float*)d_in[19];
    float* out = (float*)d_out;

    char* ws = (char*)d_ws;
    int*    chars = (int*)(ws + OFF_CHARS);
    __bf16* w1p   = (__bf16*)(ws + OFF_W1);
    __bf16* w2p   = (__bf16*)(ws + OFF_W2);
    __bf16* w3p   = (__bf16*)(ws + OFF_W3);
    __bf16* xcat  = (__bf16*)(ws + OFF_XR);
    __bf16* x1    = (__bf16*)(ws + OFF_X1);
    __bf16* x2    = (__bf16*)(ws + OFF_X2);
    __bf16* x3    = (__bf16*)(ws + OFF_X3);
    float*  yscr  = (float*)(ws + OFF_Y);

    // 0) zero halo rows of xcat (x1/x2/x3 halos are zeroed after their region
    //    becomes free, see below)
    zero_halo_kernel<<<(B_LEN * CPAD1 * 2 + 255) / 256, 256, 0, stream>>>(xcat, CPAD1);

    // 1) argmax
    argmax_kernel<<<(T_LEN * B_LEN + 255) / 256, 256, 0, stream>>>(input, chars);

    // 2) pack weights into WMMA A-fragment layout
    pack_w_kernel<<<(3 * 13 * 8 * 512 + 255) / 256, 256, 0, stream>>>(w1, w1p, 128, 400, 3, 13);
    pack_w_kernel<<<(3 * 4 * 4 * 512 + 255) / 256, 256, 0, stream>>>(w2, w2p, 64, 128, 3, 4);
    pack_w_kernel<<<(3 * 2 * 2 * 512 + 255) / 256, 256, 0, stream>>>(w3, w3p, 32, 64, 3, 2);

    // 3) build concatenated input (channel-contiguous bf16, halo-padded)
    build_x_kernel<<<(B_LEN * T_LEN * CPAD1 + 255) / 256, 256, 0, stream>>>(
        input, g_style, char_style, chars, xcat);

    dim3 gconv(T_LEN / 128, B_LEN);

    // 4) conv1 (400->128) + GN + ReLU        (x1 overlays dead xcat region)
    conv_gemm_kernel<128, 13, 2, 4><<<gconv, 256, 0, stream>>>(xcat, w1p, b1, yscr);
    zero_halo_kernel<<<(B_LEN * 128 * 2 + 255) / 256, 256, 0, stream>>>(x1, 128);
    gn_relu_kernel<128><<<B_LEN * 32, 256, 0, stream>>>(yscr, x1, g1s, g1b);

    // 5) conv2 (128->64) + GN + ReLU
    conv_gemm_kernel<64, 4, 2, 2><<<gconv, 256, 0, stream>>>(x1, w2p, b2, yscr);
    zero_halo_kernel<<<(B_LEN * 64 * 2 + 255) / 256, 256, 0, stream>>>(x2, 64);
    gn_relu_kernel<64><<<B_LEN * 32, 256, 0, stream>>>(yscr, x2, g2s, g2b);

    // 6) conv3 (64->32) + GN + ReLU
    conv_gemm_kernel<32, 2, 1, 2><<<gconv, 256, 0, stream>>>(x2, w3p, b3, yscr);
    zero_halo_kernel<<<(B_LEN * 32 * 2 + 255) / 256, 256, 0, stream>>>(x3, 32);
    gn_relu_kernel<32><<<B_LEN * 32, 256, 0, stream>>>(yscr, x3, g3s, g3b);

    // 7) conv4 (32->1, k=1) + affine -> (T,B,1)
    conv4_out_kernel<<<(T_LEN * B_LEN + 255) / 256, 256, 0, stream>>>(
        x3, w4, b4, mean, stdv, out);
}
